// MultiHeadAttention_11252814315763
// MI455X (gfx1250) — matrix operand order repaired and checked by reading, not compiled, not run on hardware
//
#include <hip/hip_runtime.h>
#include <hip/hip_bf16.h>

// ---------------------------------------------------------------------------
// MI455X (gfx1250) wave32 WMMA multi-head attention forward.
// Async global->LDS staging (ASYNCcnt), ring-buffered, + WMMA f16.
// ---------------------------------------------------------------------------

typedef __attribute__((ext_vector_type(16))) _Float16 v16h;
typedef __attribute__((ext_vector_type(8)))  _Float16 v8h;
typedef __attribute__((ext_vector_type(4)))  _Float16 v4h;
typedef __attribute__((ext_vector_type(8)))  float    v8f;

#define D_MODEL 1024
#define NHEAD   16
#define HEAD_DIM 64
#define SEQ     2048
#define BATCH   2
#define NTOK    (BATCH * SEQ)   // 4096

__device__ __forceinline__ v8f wmma_f16(v16h a, v16h b, v8f c) {
  // D = A(16x32 f16) x B(32x16 f16) + C(16x16 f32)
  return __builtin_amdgcn_wmma_f32_16x16x32_f16(
      /*neg_a=*/false, a, /*neg_b=*/false, b,
      /*c_mod=*/(short)0, c, /*reuse_a=*/false, /*reuse_b=*/false);
}

// A-matrix 16x32 f16 fragment (contiguous K order, matching B):
//   lane L (row = L%16): halves 0..15 -> K = 16*(L/16)+i
__device__ __forceinline__ v16h load_a16(const _Float16* base, int lhalf) {
  v8h lo = *(const v8h*)(base + lhalf * 16);
  v8h hi = *(const v8h*)(base + lhalf * 16 + 8);
  v16h a;
#pragma unroll
  for (int i = 0; i < 8; ++i) { a[i] = lo[i]; a[i + 8] = hi[i]; }
  return a;
}

// Low 32 bits of a generic pointer to __shared__ = LDS byte offset
// (flat->LDS aperture mapping).
__device__ __forceinline__ unsigned lds_addr_of(const void* p) {
  return (unsigned)(unsigned long long)p;
}

// Async copy 16 bytes / lane from global to LDS; tracked by ASYNCcnt.
__device__ __forceinline__ void async_b128(unsigned lds, const void* gaddr) {
  asm volatile("global_load_async_to_lds_b128 %0, %1, off"
               :: "v"(lds), "v"(gaddr) : "memory");
}

// ---------------------------------------------------------------------------
// f32 -> f16 conversion (4 elements / thread)
// ---------------------------------------------------------------------------
__global__ void mha_cvt_f16(const float* __restrict__ in,
                            _Float16* __restrict__ out, int n4) {
  int i = blockIdx.x * blockDim.x + threadIdx.x;
  if (i >= n4) return;
  float4 v = ((const float4*)in)[i];
  v4h h;
  h[0] = (_Float16)v.x; h[1] = (_Float16)v.y;
  h[2] = (_Float16)v.z; h[3] = (_Float16)v.w;
  ((v4h*)out)[i] = h;
}

// ---------------------------------------------------------------------------
// Projection GEMM: C[m,n] = X[m,:] . W[n,:] + bias[n]
// Weight B-tile (64 cols x 32 k) staged in LDS via async copy, shared by all
// 8 waves; double buffered; K-loop fully unrolled (no loop phis -> WMMA
// accumulates in place). Output f16 to [B,H,S,hd] (mode 0) or [B,H,hd,S]
// (mode 1, for V). Grid: (NTOK/128, D_MODEL/64), block 256.
// ---------------------------------------------------------------------------
__global__ void mha_proj_kernel(const _Float16* __restrict__ X,
                                const _Float16* __restrict__ W,
                                const float* __restrict__ bias,
                                _Float16* __restrict__ out, int mode) {
  __shared__ _Float16 bt[2][64 * 32];   // [n][k] row-major, 4KB per buffer

  const int tid   = threadIdx.x;
  const int wave  = tid >> 5;
  const int lane  = tid & 31;
  const int lhalf = lane >> 4;
  const int lm    = lane & 15;

  const int m0 = blockIdx.x * 128 + wave * 16;
  const int n0 = blockIdx.y * 64;

  const _Float16* arow = X + (size_t)(m0 + lm) * D_MODEL;

  // 256 threads x 16B = 4KB tile: thread t -> row n=t/4, 8-half segment t%4
  const int srow = tid >> 2, sseg = tid & 3;
  auto stageB = [&](int k0, int buf) {
    const _Float16* src = W + (size_t)(n0 + srow) * D_MODEL + k0 + sseg * 8;
    async_b128(lds_addr_of(&bt[buf][srow * 32 + sseg * 8]), src);
  };

  stageB(0, 0);
  v8f acc[4] = {};
#pragma unroll
  for (int kk = 0; kk < D_MODEL / 32; ++kk) {
    const int buf = kk & 1;
    if (kk + 1 < D_MODEL / 32) {
      stageB((kk + 1) * 32, buf ^ 1);
      asm volatile("s_wait_asynccnt 0x1" ::: "memory");
    } else {
      asm volatile("s_wait_asynccnt 0x0" ::: "memory");
    }
    __syncthreads();

    v16h a = load_a16(arow + kk * 32, lhalf);
#pragma unroll
    for (int c = 0; c < 4; ++c) {
      // B fragment: col n = c*16+lm, K = 16*(L/16)+i contiguous in LDS row
      v16h b = *(const v16h*)&bt[buf][(c * 16 + lm) * 32 + lhalf * 16];
      acc[c] = wmma_f16(a, b, acc[c]);
    }
    __syncthreads();   // protect buf before it is re-staged
  }

#pragma unroll
  for (int c = 0; c < 4; ++c) {
    const int n = n0 + c * 16 + lm;
    const float bn = bias[n];
    const int h = n >> 6;
    const int d = n & 63;
#pragma unroll
    for (int r = 0; r < 8; ++r) {
      const int m = m0 + r + lhalf * 8;
      const int b = m >> 11;
      const int s = m & 2047;
      const float val = acc[c][r] + bn;
      size_t idx;
      if (mode == 0)
        idx = (((size_t)(b * NHEAD + h) * SEQ + s) * HEAD_DIM + d);
      else
        idx = (((size_t)(b * NHEAD + h) * HEAD_DIM + d) * SEQ + s);
      out[idx] = (_Float16)val;
    }
  }
}

// ---------------------------------------------------------------------------
// Output projection GEMM -> f32, same staging scheme.
// ---------------------------------------------------------------------------
__global__ void mha_outproj_kernel(const _Float16* __restrict__ X,
                                   const _Float16* __restrict__ W,
                                   const float* __restrict__ bias,
                                   float* __restrict__ out) {
  __shared__ _Float16 bt[2][64 * 32];

  const int tid   = threadIdx.x;
  const int wave  = tid >> 5;
  const int lane  = tid & 31;
  const int lhalf = lane >> 4;
  const int lm    = lane & 15;

  const int m0 = blockIdx.x * 128 + wave * 16;
  const int n0 = blockIdx.y * 64;

  const _Float16* arow = X + (size_t)(m0 + lm) * D_MODEL;

  const int srow = tid >> 2, sseg = tid & 3;
  auto stageB = [&](int k0, int buf) {
    const _Float16* src = W + (size_t)(n0 + srow) * D_MODEL + k0 + sseg * 8;
    async_b128(lds_addr_of(&bt[buf][srow * 32 + sseg * 8]), src);
  };

  stageB(0, 0);
  v8f acc[4] = {};
#pragma unroll
  for (int kk = 0; kk < D_MODEL / 32; ++kk) {
    const int buf = kk & 1;
    if (kk + 1 < D_MODEL / 32) {
      stageB((kk + 1) * 32, buf ^ 1);
      asm volatile("s_wait_asynccnt 0x1" ::: "memory");
    } else {
      asm volatile("s_wait_asynccnt 0x0" ::: "memory");
    }
    __syncthreads();

    v16h a = load_a16(arow + kk * 32, lhalf);
#pragma unroll
    for (int c = 0; c < 4; ++c) {
      v16h b = *(const v16h*)&bt[buf][(c * 16 + lm) * 32 + lhalf * 16];
      acc[c] = wmma_f16(a, b, acc[c]);
    }
    __syncthreads();
  }

#pragma unroll
  for (int c = 0; c < 4; ++c) {
    const int n = n0 + c * 16 + lm;
    const float bn = bias[n];
#pragma unroll
    for (int r = 0; r < 8; ++r) {
      const int m = m0 + r + lhalf * 8;
      out[(size_t)m * D_MODEL + n] = acc[c][r] + bn;
    }
  }
}

// ---------------------------------------------------------------------------
// Flash attention. Block = 8 waves = 128 query rows of one (b,h).
// K tile (32 keys x 64 dims) and V tile (64 dims x 32 keys, V pre-transposed)
// staged in LDS via async copies in a 4-deep ring -> only ONE barrier per
// iteration (a buffer is rewritten 3 iters after its last read; the two
// intervening barriers order the rewrite after all reads).
// Per-wave P staging in LDS converts C-layout -> A-layout for the P.V WMMA.
// Grid = B*H*(S/128) = 512.
// ---------------------------------------------------------------------------
__global__ void mha_attn_kernel(const _Float16* __restrict__ Q,
                                const _Float16* __restrict__ K,
                                const _Float16* __restrict__ Vt,
                                _Float16* __restrict__ out) {
  __shared__ _Float16 kt[4][32 * 64];    // [key][dim]   4KB each
  __shared__ _Float16 vt[4][64 * 32];    // [dim][key]   4KB each
  __shared__ _Float16 plds[8][16 * 32];  // per-wave P staging

  const int tid   = threadIdx.x;
  const int bh    = blockIdx.x >> 4;
  const int tile  = blockIdx.x & 15;
  const int wave  = tid >> 5;
  const int lane  = tid & 31;
  const int lhalf = lane >> 4;
  const int lm    = lane & 15;

  const int b = bh >> 4;
  const int h = bh & 15;
  const int s0 = tile * 128 + wave * 16;

  const _Float16* Qb = Q  + (((size_t)bh * SEQ) + s0 + lm) * HEAD_DIM;
  const _Float16* Kb = K  + (size_t)bh * SEQ * HEAD_DIM;
  const _Float16* Vb = Vt + (size_t)bh * HEAD_DIM * SEQ;
  _Float16* pw = &plds[wave][0];

  v16h q0 = load_a16(Qb, lhalf);
  v16h q1 = load_a16(Qb + 32, lhalf);

  // K tile: contiguous 4KB -> thread t copies bytes [16t,16t+16)
  // V tile: 64 rows (dims) x 64B; thread t -> row t/4, segment t%4
  const int vrow = tid >> 2, vseg = tid & 3;
  auto stage = [&](int j0, int buf) {
    async_b128(lds_addr_of(&kt[buf][tid * 8]), Kb + (size_t)j0 * HEAD_DIM + tid * 8);
    async_b128(lds_addr_of(&vt[buf][vrow * 32 + vseg * 8]),
               Vb + (size_t)vrow * SEQ + j0 + vseg * 8);
  };

  float mrow[8], lrow[8];
#pragma unroll
  for (int r = 0; r < 8; ++r) { mrow[r] = -3.0e38f; lrow[r] = 0.0f; }
  v8f o[4] = {};

  const float SCL = 0.125f;          // 1/sqrt(64)
  const float L2E = 1.44269504f;     // log2(e)

  const int NJ = SEQ / 32;
  stage(0, 0);
#pragma unroll 4
  for (int j = 0; j < NJ; ++j) {
    const int buf = j & 3;
    // branch-free prefetch of next tile (clamped redundant re-stage at end)
    int jn = j + 1; if (jn >= NJ) jn = NJ - 1;
    stage(jn * 32, (j + 1) & 3);
    asm volatile("s_wait_asynccnt 0x2" ::: "memory");  // tile j's 2 asyncs done
    __syncthreads();

    // ---- scores: two 16x16 tiles from LDS K tile ----
    v8f sa = {}, sb = {};
    {
      const _Float16* kp = &kt[buf][lm * HEAD_DIM + lhalf * 16];
      v16h b0 = *(const v16h*)kp;
      v16h b1 = *(const v16h*)(kp + 32);
      sa = wmma_f16(q0, b0, sa);
      sa = wmma_f16(q1, b1, sa);
    }
    {
      const _Float16* kp = &kt[buf][(16 + lm) * HEAD_DIM + lhalf * 16];
      v16h b0 = *(const v16h*)kp;
      v16h b1 = *(const v16h*)(kp + 32);
      sb = wmma_f16(q0, b0, sb);
      sb = wmma_f16(q1, b1, sb);
    }

    // ---- online softmax ----
#pragma unroll
    for (int r = 0; r < 8; ++r) {
      float x0 = sa[r] * SCL;
      float x1 = sb[r] * SCL;
      float t = fmaxf(x0, x1);
      t = fmaxf(t, __shfl_xor(t, 1, 32));
      t = fmaxf(t, __shfl_xor(t, 2, 32));
      t = fmaxf(t, __shfl_xor(t, 4, 32));
      t = fmaxf(t, __shfl_xor(t, 8, 32));
      const float mn = fmaxf(mrow[r], t);
      const float al = exp2f((mrow[r] - mn) * L2E);
      const float p0 = exp2f((x0 - mn) * L2E);
      const float p1 = exp2f((x1 - mn) * L2E);
      float rs = p0 + p1;
      rs += __shfl_xor(rs, 1, 32);
      rs += __shfl_xor(rs, 2, 32);
      rs += __shfl_xor(rs, 4, 32);
      rs += __shfl_xor(rs, 8, 32);
      lrow[r] = lrow[r] * al + rs;
      mrow[r] = mn;
#pragma unroll
      for (int c = 0; c < 4; ++c) o[c][r] *= al;
      const int row = r + lhalf * 8;
      pw[row * 32 + lm]      = (_Float16)p0;
      pw[row * 32 + 16 + lm] = (_Float16)p1;
    }

    // cross-lane LDS RAW within the wave -> explicit DS wait
    asm volatile("s_wait_dscnt 0x0" ::: "memory");

    v16h pf;
    {
      v8h lo = *(const v8h*)(pw + lm * 32 + lhalf * 16);
      v8h hi = *(const v8h*)(pw + lm * 32 + lhalf * 16 + 8);
#pragma unroll
      for (int i = 0; i < 8; ++i) { pf[i] = lo[i]; pf[i + 8] = hi[i]; }
    }

    // ---- O += P.V from LDS V tile ----
#pragma unroll
    for (int c = 0; c < 4; ++c) {
      v16h vf = *(const v16h*)&vt[buf][(c * 16 + lm) * 32 + lhalf * 16];
      o[c] = wmma_f16(pf, vf, o[c]);
    }
  }

  // ---- normalize and write out f16 [B,S,D] ----
#pragma unroll
  for (int r = 0; r < 8; ++r) {
    const float inv = 1.0f / lrow[r];
    const int token = s0 + r + lhalf * 8;
#pragma unroll
    for (int c = 0; c < 4; ++c) {
      const int col = h * HEAD_DIM + c * 16 + lm;
      out[((size_t)b * SEQ + token) * D_MODEL + col] = (_Float16)(o[c][r] * inv);
    }
  }
}

// ---------------------------------------------------------------------------
extern "C" void kernel_launch(void* const* d_in, const int* in_sizes, int n_in,
                              void* d_out, int out_size, void* d_ws, size_t ws_size,
                              hipStream_t stream) {
  (void)in_sizes; (void)n_in; (void)out_size; (void)ws_size;

  const float* q_in = (const float*)d_in[0];
  const float* k_in = (const float*)d_in[1];
  const float* v_in = (const float*)d_in[2];
  const float* Wq   = (const float*)d_in[3];
  const float* bq   = (const float*)d_in[4];
  const float* Wk   = (const float*)d_in[5];
  const float* bk   = (const float*)d_in[6];
  const float* Wv   = (const float*)d_in[7];
  const float* bv   = (const float*)d_in[8];
  const float* Wo   = (const float*)d_in[9];
  const float* bo   = (const float*)d_in[10];
  float* out = (float*)d_out;

  const size_t NE = (size_t)NTOK * D_MODEL;      // 4M elems
  const size_t WE = (size_t)D_MODEL * D_MODEL;   // 1M elems

  _Float16* p = (_Float16*)d_ws;
  _Float16* Xq16 = p; p += NE;
  _Float16* Xk16 = p; p += NE;
  _Float16* Xv16 = p; p += NE;
  _Float16* Wq16 = p; p += WE;
  _Float16* Wk16 = p; p += WE;
  _Float16* Wv16 = p; p += WE;
  _Float16* Wo16 = p; p += WE;
  _Float16* Qb   = p; p += NE;   // [B,H,S,hd]
  _Float16* Kb   = p; p += NE;   // [B,H,S,hd]
  _Float16* Vtb  = p; p += NE;   // [B,H,hd,S]
  _Float16* A16  = p; p += NE;   // attention output [B,S,D] f16

  const int n4x = (int)(NE / 4), n4w = (int)(WE / 4);
  hipLaunchKernelGGL(mha_cvt_f16, dim3((n4x + 255) / 256), dim3(256), 0, stream, q_in, Xq16, n4x);
  hipLaunchKernelGGL(mha_cvt_f16, dim3((n4x + 255) / 256), dim3(256), 0, stream, k_in, Xk16, n4x);
  hipLaunchKernelGGL(mha_cvt_f16, dim3((n4x + 255) / 256), dim3(256), 0, stream, v_in, Xv16, n4x);
  hipLaunchKernelGGL(mha_cvt_f16, dim3((n4w + 255) / 256), dim3(256), 0, stream, Wq, Wq16, n4w);
  hipLaunchKernelGGL(mha_cvt_f16, dim3((n4w + 255) / 256), dim3(256), 0, stream, Wk, Wk16, n4w);
  hipLaunchKernelGGL(mha_cvt_f16, dim3((n4w + 255) / 256), dim3(256), 0, stream, Wv, Wv16, n4w);
  hipLaunchKernelGGL(mha_cvt_f16, dim3((n4w + 255) / 256), dim3(256), 0, stream, Wo, Wo16, n4w);

  dim3 pgrid(NTOK / 128, D_MODEL / 64);
  hipLaunchKernelGGL(mha_proj_kernel, pgrid, dim3(256), 0, stream, Xq16, Wq16, bq, Qb, 0);
  hipLaunchKernelGGL(mha_proj_kernel, pgrid, dim3(256), 0, stream, Xk16, Wk16, bk, Kb, 0);
  hipLaunchKernelGGL(mha_proj_kernel, pgrid, dim3(256), 0, stream, Xv16, Wv16, bv, Vtb, 1);

  hipLaunchKernelGGL(mha_attn_kernel, dim3(BATCH * NHEAD * (SEQ / 128)), dim3(256), 0, stream,
                     Qb, Kb, Vtb, A16);

  hipLaunchKernelGGL(mha_outproj_kernel, pgrid, dim3(256), 0, stream, A16, Wo16, bo, out);
}
